// GATNet_33131377721796
// MI455X (gfx1250) — matrix operand order repaired and checked by reading, not compiled, not run on hardware
//
#include <hip/hip_runtime.h>
#include <hip/hip_bf16.h>

typedef __attribute__((ext_vector_type(16))) _Float16 v16h;
typedef __attribute__((ext_vector_type(8)))  float    v8f;

#define NEG_SLOPE 0.2f
#define TPB 256

// ---------- float <-> monotone-ordered uint (for atomicMax on floats) ----------
__device__ __forceinline__ unsigned ordf(float f) {
  unsigned u = __float_as_uint(f);
  return (u >> 31) ? ~u : (u | 0x80000000u);
}
__device__ __forceinline__ float unordf(unsigned o) {
  return (o >> 31) ? __uint_as_float(o & 0x7fffffffu) : __uint_as_float(~o);
}

// ---------- conversion kernels ----------
__global__ void k_f32_to_f16(const float* __restrict__ in, _Float16* __restrict__ out, long n) {
  long i = (long)blockIdx.x * blockDim.x + threadIdx.x;
  if (i < n) out[i] = (_Float16)in[i];
}

// W [K][Fout] fp32 -> Wt [Fout][K] f16 (so B-fragment K-pairs are contiguous)
__global__ void k_transposeW(const float* __restrict__ W, _Float16* __restrict__ Wt, int K, int Fout) {
  int i = blockIdx.x * blockDim.x + threadIdx.x;
  if (i < K * Fout) {
    int k = i / Fout, n = i % Fout;
    Wt[n * K + k] = (_Float16)W[i];
  }
}

// ---------- WMMA GEMM: C[N,Fout] = A16[N,K] * W (Bt is [Fout][K] f16) ----------
// Each wave: one 16-row tile x 64 output cols (4 WMMA accumulators). 8 waves/block.
__global__ __launch_bounds__(256) void k_gemm_wmma(const _Float16* __restrict__ A,
                                                   const _Float16* __restrict__ Bt,
                                                   float* __restrict__ C,
                                                   int rowTiles, int K, int Fout) {
  const int lane = threadIdx.x & 31;
  const int wave = threadIdx.x >> 5;
  const int colGroups = Fout >> 6;                    // 64-col groups
  const int cg = blockIdx.x % colGroups;
  const int rt = (blockIdx.x / colGroups) * 8 + wave; // row tile
  if (rt >= rowTiles) return;                         // wave-uniform exit (EXEC all-1 inside)
  const int l15 = lane & 15;
  const int lh  = lane >> 4;                          // lane half (K-group select)
  const _Float16* arow = A + (long)(rt * 16 + l15) * K;
  v8f acc[4] = {};
  const int ncb = cg * 64;

  for (int kb = 0; kb < K; kb += 32) {
    // A fragment: 16x32 f16. lane = row M; VGPR r holds K pair per ISA layout.
    union { v16h v; unsigned u[8]; } af;
#pragma unroll
    for (int r = 0; r < 8; ++r) {
      int koff = ((r < 4) ? 0 : 16) + lh * 8 + 2 * (r & 3);
      af.u[r] = *(const unsigned*)(arow + kb + koff);
    }
#pragma unroll
    for (int t = 0; t < 4; ++t) {
      // B fragment: 32x16 f16. lane = col N; K pairs contiguous in Wt row.
      const _Float16* brow = Bt + (long)(ncb + t * 16 + l15) * K + kb;
      union { v16h v; unsigned u[8]; } bf;
#pragma unroll
      for (int r = 0; r < 8; ++r) bf.u[r] = *(const unsigned*)(brow + ((r < 4) ? 0 : 16) + lh * 8 + 2 * (r & 3));
      acc[t] = __builtin_amdgcn_wmma_f32_16x16x32_f16(
          /*neg_a=*/false, af.v, /*neg_b=*/false, bf.v,
          /*c_mod=*/(short)0, acc[t], /*reuse_a=*/false, /*reuse_b=*/false);
    }
  }
  // C/D layout: lanes 0-15 -> M=0..7 (VGPR idx), lanes 16-31 -> M=8..15; N = lane&15
#pragma unroll
  for (int t = 0; t < 4; ++t) {
    int ncol = ncb + t * 16 + l15;
#pragma unroll
    for (int r = 0; r < 8; ++r) {
      int mrow = lh * 8 + r;
      C[(long)(rt * 16 + mrow) * Fout + ncol] = acc[t][r];
    }
  }
}

// ---------- attention scores: e_src/e_dst [N,8] ----------
__global__ void k_scores(const float* __restrict__ Hm, const float* __restrict__ as_,
                         const float* __restrict__ ad_, float* __restrict__ es,
                         float* __restrict__ ed, int N, int C) {
  int i = blockIdx.x * blockDim.x + threadIdx.x;
  if (i >= N * 8) return;
  int n = i >> 3, h = i & 7;
  const float* hp = Hm + (long)n * 8 * C + h * C;
  const float* a1 = as_ + h * C;
  const float* a2 = ad_ + h * C;
  float s1 = 0.f, s2 = 0.f;
  for (int c = 0; c < C; ++c) { float v = hp[c]; s1 += v * a1[c]; s2 += v * a2[c]; }
  es[i] = s1; ed[i] = s2;
}

// ---------- per-layer init: O=0, m=ord(-inf), denom=0 ----------
__global__ void k_init(float* __restrict__ O, long osz, unsigned* __restrict__ m,
                       float* __restrict__ dn, long nh) {
  long i = (long)blockIdx.x * blockDim.x + threadIdx.x;
  if (i < osz) O[i] = 0.f;
  if (i < nh) { m[i] = 0x007FFFFFu; dn[i] = 0.f; }  // ord(-inf) = ~0xFF800000
}

__device__ __forceinline__ void edge_sd(const int* __restrict__ ei, long E, long e, int& s, int& d) {
  if (e < E) { s = ei[e]; d = ei[E + e]; }
  else       { s = (int)(e - E); d = s; }            // self loops appended
}

// ---------- edge pass A: segment max ----------
__global__ void k_edge_max(const int* __restrict__ ei, long E, long Etot,
                           const float* __restrict__ es, const float* __restrict__ ed,
                           unsigned* __restrict__ m) {
  long i = (long)blockIdx.x * blockDim.x + threadIdx.x;
  if (i >= Etot * 8) return;
  long e = i >> 3; int h = (int)(i & 7);
  int s, d; edge_sd(ei, E, e, s, d);
  float ev = es[(long)s * 8 + h] + ed[(long)d * 8 + h];
  ev = ev > 0.f ? ev : NEG_SLOPE * ev;
  atomicMax(&m[(long)d * 8 + h], ordf(ev));
}

// ---------- edge pass B: w = exp(e - m[dst]); denom += w ----------
__global__ void k_edge_sum(const int* __restrict__ ei, long E, long Etot,
                           const float* __restrict__ es, const float* __restrict__ ed,
                           const unsigned* __restrict__ m, float* __restrict__ dn,
                           float* __restrict__ wedge) {
  long i = (long)blockIdx.x * blockDim.x + threadIdx.x;
  if (i >= Etot * 8) return;
  long e = i >> 3; int h = (int)(i & 7);
  int s, d; edge_sd(ei, E, e, s, d);
  float ev = es[(long)s * 8 + h] + ed[(long)d * 8 + h];
  ev = ev > 0.f ? ev : NEG_SLOPE * ev;
  float w = __expf(ev - unordf(m[(long)d * 8 + h]));
  wedge[i] = w;
  atomicAdd(&dn[(long)d * 8 + h], w);
}

// ---------- edge pass C: O[dst,:] += H[src,:] * alpha ----------
__global__ void k_edge_agg(const int* __restrict__ ei, long E, long Etot,
                           const float* __restrict__ wedge, const float* __restrict__ dn,
                           const float* __restrict__ Hm, float* __restrict__ O,
                           int HC, int C) {
  long i = (long)blockIdx.x * blockDim.x + threadIdx.x;
  if (i >= Etot * HC) return;
  long e = i / HC; int j = (int)(i % HC); int h = j / C;
  int s, d; edge_sd(ei, E, e, s, d);
  float alpha = wedge[e * 8 + h] / (dn[(long)d * 8 + h] + 1e-16f);
  atomicAdd(&O[(long)d * HC + j], Hm[(long)s * HC + j] * alpha);
}

// ---------- epilogue layers 1-2: ELU(O + b) -> next-layer f16 input ----------
__global__ void k_elu_to_f16(const float* __restrict__ O, const float* __restrict__ b,
                             _Float16* __restrict__ X16, long n, int HC) {
  long i = (long)blockIdx.x * blockDim.x + threadIdx.x;
  if (i < n) {
    float v = O[i] + b[(int)(i % HC)];
    v = v > 0.f ? v : (__expf(v) - 1.f);
    X16[i] = (_Float16)v;
  }
}

// ---------- epilogue layer 3: mean over heads + b3 -> d_out ----------
__global__ void k_mean_out(const float* __restrict__ O, const float* __restrict__ b3,
                           float* __restrict__ out, int N, int Hh, int C) {
  int i = blockIdx.x * blockDim.x + threadIdx.x;
  if (i >= N * C) return;
  int n = i / C, c = i % C;
  float s = 0.f;
  for (int h = 0; h < Hh; ++h) s += O[(long)n * Hh * C + h * C + c];
  out[i] = s / (float)Hh + b3[c];
}

static inline unsigned gridFor(long n) { return (unsigned)((n + TPB - 1) / TPB); }

extern "C" void kernel_launch(void* const* d_in, const int* in_sizes, int n_in,
                              void* d_out, int out_size, void* d_ws, size_t ws_size,
                              hipStream_t stream) {
  const float* x   = (const float*)d_in[0];
  const int*   ei  = (const int*)d_in[1];
  const float* W1  = (const float*)d_in[2];
  const float* a1s = (const float*)d_in[3];
  const float* a1d = (const float*)d_in[4];
  const float* b1  = (const float*)d_in[5];
  const float* W2  = (const float*)d_in[6];
  const float* a2s = (const float*)d_in[7];
  const float* a2d = (const float*)d_in[8];
  const float* b2  = (const float*)d_in[9];
  const float* W3  = (const float*)d_in[10];
  const float* a3s = (const float*)d_in[11];
  const float* a3d = (const float*)d_in[12];
  const float* b3  = (const float*)d_in[13];

  const int K    = 128;                 // F_in = H*HID = 128 for all layers
  const int N    = in_sizes[0] / K;     // 50000
  const long E   = in_sizes[1] / 2;     // 800000
  const long Etot = E + N;              // + self loops
  const int rowTiles = (N + 15) / 16;   // 3125
  const int rowBlocks = (rowTiles + 7) / 8;

  // ---- workspace carving ----
  char* ws = (char*)d_ws;
  auto carve = [&](size_t bytes) { void* p = (void*)ws; ws += (bytes + 255) & ~(size_t)255; return p; };
  _Float16* X16   = (_Float16*)carve((size_t)N * K * sizeof(_Float16));
  _Float16* Wt    = (_Float16*)carve((size_t)320 * K * sizeof(_Float16));
  float*    Hm    = (float*)carve((size_t)N * 320 * sizeof(float));
  float*    O     = (float*)carve((size_t)N * 320 * sizeof(float));
  float*    es    = (float*)carve((size_t)N * 8 * sizeof(float));
  float*    ed    = (float*)carve((size_t)N * 8 * sizeof(float));
  unsigned* m     = (unsigned*)carve((size_t)N * 8 * sizeof(unsigned));
  float*    dn    = (float*)carve((size_t)N * 8 * sizeof(float));
  float*    wedge = (float*)carve((size_t)Etot * 8 * sizeof(float));

  // x -> f16 input for layer 1 GEMM
  k_f32_to_f16<<<gridFor((long)N * K), TPB, 0, stream>>>(x, X16, (long)N * K);

  auto run_layer = [&](const float* W, const float* as_, const float* ad_,
                       int Fout, int C) {
    k_transposeW<<<gridFor((long)K * Fout), TPB, 0, stream>>>(W, Wt, K, Fout);
    k_gemm_wmma<<<dim3((unsigned)(rowBlocks * (Fout / 64))), TPB, 0, stream>>>(
        X16, Wt, Hm, rowTiles, K, Fout);
    k_scores<<<gridFor((long)N * 8), TPB, 0, stream>>>(Hm, as_, ad_, es, ed, N, C);
    k_init<<<gridFor((long)N * Fout), TPB, 0, stream>>>(O, (long)N * Fout, m, dn, (long)N * 8);
    k_edge_max<<<gridFor(Etot * 8), TPB, 0, stream>>>(ei, E, Etot, es, ed, m);
    k_edge_sum<<<gridFor(Etot * 8), TPB, 0, stream>>>(ei, E, Etot, es, ed, m, dn, wedge);
    k_edge_agg<<<gridFor(Etot * Fout), TPB, 0, stream>>>(ei, E, Etot, wedge, dn, Hm, O, Fout, C);
  };

  // Layer 1: Fout = 128 (8 heads x 16), concat + ELU
  run_layer(W1, a1s, a1d, 128, 16);
  k_elu_to_f16<<<gridFor((long)N * 128), TPB, 0, stream>>>(O, b1, X16, (long)N * 128, 128);

  // Layer 2: Fout = 128, concat + ELU
  run_layer(W2, a2s, a2d, 128, 16);
  k_elu_to_f16<<<gridFor((long)N * 128), TPB, 0, stream>>>(O, b2, X16, (long)N * 128, 128);

  // Layer 3: Fout = 320 (8 heads x 40), mean over heads + b3
  run_layer(W3, a3s, a3d, 320, 40);
  k_mean_out<<<gridFor((long)N * 40), TPB, 0, stream>>>(O, b3, (float*)d_out, N, 8, 40);
}
